// TiSANCR_84774064488743
// MI455X (gfx1250) — compile-verified, matrix operand order checked by reading, not run on hardware
//
#include <hip/hip_runtime.h>
#include <hip/hip_bf16.h>

// ---------------- types / constants ----------------
typedef __attribute__((ext_vector_type(16))) _Float16 v16h;
typedef __attribute__((ext_vector_type(8)))  float    v8f;
typedef __attribute__((ext_vector_type(4)))  unsigned int v4u;

#define E64    64
#define BATCH  1024
#define HL     100
#define MH     (BATCH * HL)   // 102400 history tokens
#define MS2    (BATCH * 2)    // 2048 scoring tokens (pos+neg)
#define NHEAD  4
#define SPAD   112            // 100 padded to 7*16
#define SSLD   116            // f32 score row stride (pad vs bank conflicts)
#define ATTN_SMEM ((SPAD*128*2)*2 + 64*128*2 + SPAD*SSLD*4)  // 125,696 B

union FragU { v16h h; v4u u[2]; };

// Load a 16x32 f16 fragment (A layout) from row-major [row][ld] storage.
// Per CDNA5 ISA: lanes 0-15 row M=lane, K 0..7 & 16..23; lanes 16-31 same rows,
// K 8..15 & 24..31.  B fragments use the mirrored layout with "row" == N (so we
// store weights transposed [N][K] and reuse this loader).
__device__ __forceinline__ v16h load_frag(const _Float16* base, int ld, int row0,
                                          int k0, int lane) {
  int r  = row0 + (lane & 15);
  int kb = k0 + ((lane >> 4) << 3);
  const _Float16* p = base + (size_t)r * ld + kb;
  FragU f;
  f.u[0] = *reinterpret_cast<const v4u*>(p);
  f.u[1] = *reinterpret_cast<const v4u*>(p + 16);
  return f.h;
}

// ---------------- generic WMMA GEMM: C = act(A[M,K] @ W[K,N] + bias) ----------------
// WT = W transposed, f16 [N][K] row-major.
// Each wave computes a 32x64 output block (2 M-frags x 4 N-frags, 8 accumulators).
// All 6 fragment loads of a k-step are materialized in distinct registers first so
// the scheduler can clause the 12 b128 loads, wait once, then fire 8 back-to-back
// WMMAs.  Requires M % 32 == 0 and N % 64 == 0 (true for all call sites).
__global__ void wmma_gemm_kernel(const _Float16* __restrict__ A, int lda,
                                 const _Float16* __restrict__ WT,
                                 const float* __restrict__ bias,
                                 _Float16* __restrict__ Ch, float* __restrict__ Cf,
                                 int ldc, int M, int N, int K, int relu)
{
  int lane = threadIdx.x & 31;
  int wave = threadIdx.x >> 5;
  int tile = blockIdx.x * (blockDim.x >> 5) + wave;
  int ntn = N >> 6;                          // 64-wide N blocks
  int total = (M >> 5) * ntn;
  if (tile >= total) return;                 // wave-uniform exit: EXEC stays full
  int mt = tile / ntn, nt = tile - mt * ntn;
  v8f acc[2][4] = {};
  for (int k0 = 0; k0 < K; k0 += 32) {
    v16h a0 = load_frag(A, lda, (mt << 5), k0, lane);
    v16h a1 = load_frag(A, lda, (mt << 5) + 16, k0, lane);
    v16h bfr[4];
#pragma unroll
    for (int jn = 0; jn < 4; ++jn)
      bfr[jn] = load_frag(WT, K, (nt << 6) + (jn << 4), k0, lane);
#pragma unroll
    for (int jn = 0; jn < 4; ++jn) {
      acc[0][jn] = __builtin_amdgcn_wmma_f32_16x16x32_f16(false, a0, false, bfr[jn],
                                                          (short)0, acc[0][jn], false, false);
      acc[1][jn] = __builtin_amdgcn_wmma_f32_16x16x32_f16(false, a1, false, bfr[jn],
                                                          (short)0, acc[1][jn], false, false);
    }
  }
#pragma unroll
  for (int im = 0; im < 2; ++im) {
#pragma unroll
    for (int jn = 0; jn < 4; ++jn) {
      int n  = (nt << 6) + (jn << 4) + (lane & 15);
      float bv = bias ? bias[n] : 0.0f;
      int rb = (mt << 5) + (im << 4) + ((lane >> 4) << 3);  // lanes 16-31 -> rows +8
#pragma unroll
      for (int j = 0; j < 8; ++j) {
        float v = acc[im][jn][j] + bv;
        if (relu) v = fmaxf(v, 0.0f);
        size_t idx = (size_t)(rb + j) * ldc + n;
        if (Ch) Ch[idx] = (_Float16)v; else Cf[idx] = v;
      }
    }
  }
}

// ---------------- fused attention: one workgroup per (batch, head) ----------------
// scores = (Q.K + QP.KP)/8 computed as 128-wide concat dot; softmax in LDS; ctx = P.V
__global__ void attn_fused_kernel(const _Float16* __restrict__ QC,
                                  const _Float16* __restrict__ KC,
                                  const _Float16* __restrict__ V,
                                  _Float16* __restrict__ CTX)
{
  extern __shared__ char smem[];
  _Float16* sQ  = (_Float16*)smem;            // [112][128], later reused as P
  _Float16* sK  = sQ + SPAD * 128;            // [112][128]
  _Float16* sVt = sK + SPAD * 128;            // [64][128]  V transposed [n][t]
  float*    sS  = (float*)(sVt + 64 * 128);   // [112][116]
  int b = blockIdx.x >> 2;
  int h = blockIdx.x & 3;
  int tid = threadIdx.x;
  size_t tb = (size_t)b * HL;
  for (int i = tid; i < SPAD * 128; i += 256) {
    int s = i >> 7, c = i & 127;
    _Float16 q = (_Float16)0.0f, k = (_Float16)0.0f;
    if (s < HL) {
      size_t g = (tb + s) * 512 + (size_t)h * 128 + c;
      q = QC[g]; k = KC[g];
    }
    sQ[i] = q; sK[i] = k;
  }
  for (int i = tid; i < 64 * 128; i += 256) {
    int n = i >> 7, t = i & 127;
    _Float16 v = (_Float16)0.0f;
    if (t < HL) v = V[(tb + t) * 256 + (size_t)h * 64 + n];
    sVt[i] = v;
  }
  __syncthreads();
  int lane = tid & 31, wave = tid >> 5;
  for (int t = wave; t < 49; t += 8) {        // 7x7 score tiles, K=128
    int mt = t / 7, nt = t - mt * 7;
    v8f acc = {};
    for (int k0 = 0; k0 < 128; k0 += 32) {
      v16h a  = load_frag(sQ, 128, mt << 4, k0, lane);
      v16h bm = load_frag(sK, 128, nt << 4, k0, lane);
      acc = __builtin_amdgcn_wmma_f32_16x16x32_f16(false, a, false, bm, (short)0, acc,
                                                   false, false);
    }
    int n = (nt << 4) + (lane & 15);
    int rb = (mt << 4) + ((lane >> 4) << 3);
#pragma unroll
    for (int j = 0; j < 8; ++j) sS[(rb + j) * SSLD + n] = acc[j];
  }
  __syncthreads();
  _Float16* sP = sQ;                          // reuse Q region for probabilities
  if (tid < HL) {
    const float scale = 0.125f;               // 1/sqrt(64)
    float mx = -3.0e38f;
    for (int t = 0; t < HL; ++t) mx = fmaxf(mx, sS[tid * SSLD + t] * scale);
    float sum = 0.0f;
    for (int t = 0; t < HL; ++t) {
      float e = expf(sS[tid * SSLD + t] * scale - mx);
      sS[tid * SSLD + t] = e; sum += e;
    }
    float inv = 1.0f / sum;
    for (int t = 0; t < HL; ++t)  sP[tid * 128 + t] = (_Float16)(sS[tid * SSLD + t] * inv);
    for (int t = HL; t < 128; ++t) sP[tid * 128 + t] = (_Float16)0.0f;
  } else if (tid < SPAD) {
    for (int t = 0; t < 128; ++t) sP[tid * 128 + t] = (_Float16)0.0f;
  }
  __syncthreads();
  for (int t = wave; t < 28; t += 8) {        // 7x4 ctx tiles, K=128 (zero-padded)
    int mt = t >> 2, nt = t & 3;
    v8f acc = {};
    for (int k0 = 0; k0 < 128; k0 += 32) {
      v16h a  = load_frag(sP, 128, mt << 4, k0, lane);
      v16h bm = load_frag(sVt, 128, nt << 4, k0, lane);
      acc = __builtin_amdgcn_wmma_f32_16x16x32_f16(false, a, false, bm, (short)0, acc,
                                                   false, false);
    }
    int n = (nt << 4) + (lane & 15);
    int rb = (mt << 4) + ((lane >> 4) << 3);
#pragma unroll
    for (int j = 0; j < 8; ++j) {
      int s = rb + j;
      if (s < HL) CTX[(tb + s) * 256 + (size_t)h * 64 + n] = (_Float16)acc[j];
    }
  }
}

// ---------------- sequential logic-OR scan, parallel over batch ----------------
// 16 batch rows per workgroup, 4 waves = 4 N-tiles. 99 sequential MLP2 steps.
// Each step's 16x64 f16 x-tile is fetched memory->LDS with the CDNA5 async DMA
// path (GLOBAL_LOAD_ASYNC_TO_LDS_B128, tracked by ASYNCcnt).
__global__ void orscan_kernel(const _Float16* __restrict__ EV,
                              const _Float16* __restrict__ W1T, const float* __restrict__ b1,
                              const _Float16* __restrict__ W2T, const float* __restrict__ b2,
                              _Float16* __restrict__ ORV)
{
  __shared__ __align__(16) _Float16 sW1[64 * 128];
  __shared__ __align__(16) _Float16 sW2[64 * 64];
  __shared__ __align__(16) _Float16 sA[16 * 128];   // [carry | x]
  __shared__ __align__(16) _Float16 sH[16 * 64];
  int tid = threadIdx.x;                      // 128 threads
  size_t b0 = (size_t)blockIdx.x * 16;
  for (int i = tid; i < 64 * 128; i += 128) sW1[i] = W1T[i];
  for (int i = tid; i < 64 * 64; i += 128)  sW2[i] = W2T[i];
  for (int i = tid; i < 16 * 64; i += 128) {
    int r = i >> 6, c = i & 63;
    sA[r * 128 + c] = EV[((b0 + r) * HL) * 64 + c];     // carry = ev[:,0,:]
  }
  __syncthreads();
  int lane = tid & 31, wave = tid >> 5;       // wave == N-tile (0..3)
  for (int hh = 1; hh < HL; ++hh) {
    // async DMA: 32x64 f16 = 256 x 16B chunks, 2 per thread
#pragma unroll
    for (int cc = 0; cc < 2; ++cc) {
      int c = tid + cc * 128;
      int r = c >> 3, col = (c & 7) << 3;
      unsigned ldsoff = (unsigned)(size_t)(void*)&sA[r * 128 + 64 + col];
      const _Float16* g = EV + ((b0 + r) * HL + hh) * 64 + col;
      asm volatile("global_load_async_to_lds_b128 %0, %1, off"
                   :: "v"(ldsoff), "v"(g) : "memory");
    }
#if __has_builtin(__builtin_amdgcn_s_wait_asynccnt)
    __builtin_amdgcn_s_wait_asynccnt(0);
#else
    asm volatile("s_wait_asynccnt 0x0" ::: "memory");
#endif
    __syncthreads();
    v8f acc = {};
    for (int k0 = 0; k0 < 128; k0 += 32) {
      v16h a  = load_frag(sA, 128, 0, k0, lane);
      v16h bm = load_frag(sW1, 128, wave << 4, k0, lane);
      acc = __builtin_amdgcn_wmma_f32_16x16x32_f16(false, a, false, bm, (short)0, acc,
                                                   false, false);
    }
    {
      int n = (wave << 4) + (lane & 15);
      float bv = b1[n];
      int rb = (lane >> 4) << 3;
#pragma unroll
      for (int j = 0; j < 8; ++j) sH[(rb + j) * 64 + n] = (_Float16)fmaxf(acc[j] + bv, 0.0f);
    }
    __syncthreads();
    v8f acc2 = {};
    for (int k0 = 0; k0 < 64; k0 += 32) {
      v16h a  = load_frag(sH, 64, 0, k0, lane);
      v16h bm = load_frag(sW2, 64, wave << 4, k0, lane);
      acc2 = __builtin_amdgcn_wmma_f32_16x16x32_f16(false, a, false, bm, (short)0, acc2,
                                                    false, false);
    }
    {
      int n = (wave << 4) + (lane & 15);
      float bv = b2[n];
      int rb = (lane >> 4) << 3;
#pragma unroll
      for (int j = 0; j < 8; ++j) sA[(rb + j) * 128 + n] = (_Float16)(acc2[j] + bv);
    }
    __syncthreads();
  }
  for (int i = tid; i < 16 * 64; i += 128) {
    int r = i >> 6, c = i & 63;
    ORV[(b0 + r) * 64 + c] = sA[r * 128 + c];
  }
}

// ---------------- small helper kernels ----------------
__global__ void transpose_w_kernel(const float* __restrict__ W, _Float16* __restrict__ WT,
                                   int K, int N) {
  int i = blockIdx.x * blockDim.x + threadIdx.x;
  if (i >= K * N) return;
  int n = i / K, k = i - n * K;
  WT[(size_t)n * K + k] = (_Float16)W[(size_t)k * N + n];
}

// Combined [WQ;0 | 0;WQP] -> transposed f16 [512][128]; combined bias f32 [512]
__global__ void build_qkc_kernel(const float* __restrict__ Wm, const float* __restrict__ Wp,
                                 const float* __restrict__ bm, const float* __restrict__ bp,
                                 _Float16* __restrict__ WT, float* __restrict__ bc) {
  int i = blockIdx.x * blockDim.x + threadIdx.x;
  if (i < 512 * 128) {
    int n = i >> 7, k = i & 127;
    int h = n >> 7, j = n & 127;
    float v = 0.0f;
    if (j < 64) { if (k < 64)  v = Wm[(size_t)k * 256 + h * 64 + j]; }
    else        { if (k >= 64) v = Wp[(size_t)(k - 64) * 256 + h * 64 + (j - 64)]; }
    WT[(size_t)n * 128 + k] = (_Float16)v;
  }
  if (i < 512) {
    int h = i >> 7, j = i & 127;
    bc[i] = (j < 64) ? bm[h * 64 + j] : bp[h * 64 + (j - 64)];
  }
}

__global__ void build_vc_kernel(const float* __restrict__ WV, _Float16* __restrict__ WT) {
  int i = blockIdx.x * blockDim.x + threadIdx.x;
  if (i >= 256 * 128) return;
  int n = i >> 7, k = i & 127;
  WT[(size_t)n * 128 + k] = (_Float16)((k < 64) ? WV[(size_t)k * 256 + n] : 0.0f);
}

__global__ void gather_hist_kernel(const int* __restrict__ uid, const int* __restrict__ hist,
                                   const int* __restrict__ hts,
                                   const float* __restrict__ UE, const float* __restrict__ IE,
                                   const float* __restrict__ TE,
                                   _Float16* __restrict__ A1, _Float16* __restrict__ A2,
                                   _Float16* __restrict__ XH) {
  int i = blockIdx.x * blockDim.x + threadIdx.x;
  if (i >= MH * 64) return;
  int e = i & 63, m = i >> 6;
  int b = m / HL;
  size_t row = (size_t)m * 128;
  A1[row + e]      = (_Float16)UE[(size_t)uid[b] * 64 + e];
  A1[row + 64 + e] = (_Float16)IE[(size_t)hist[m] * 64 + e];
  _Float16 t = (_Float16)TE[(size_t)hts[m] * 64 + e];
  A2[row + 64 + e] = t;
  XH[row + 64 + e] = t;
}

// residual + layernorm, one wave per token row
__global__ void ln_kernel(const float* __restrict__ O, const _Float16* __restrict__ XH,
                          const float* __restrict__ g, const float* __restrict__ bta,
                          _Float16* __restrict__ AO, int M) {
  int gw = (blockIdx.x * blockDim.x + threadIdx.x) >> 5;
  int lane = threadIdx.x & 31;
  if (gw >= M) return;
  size_t ro = (size_t)gw * 64;
  float v0 = O[ro + lane]      + (float)XH[(size_t)gw * 128 + lane];
  float v1 = O[ro + 32 + lane] + (float)XH[(size_t)gw * 128 + 32 + lane];
  float s = v0 + v1, ss = v0 * v0 + v1 * v1;
  for (int m = 16; m; m >>= 1) { s += __shfl_xor(s, m, 32); ss += __shfl_xor(ss, m, 32); }
  float mean = s * 0.015625f;
  float var  = ss * 0.015625f - mean * mean;
  float inv  = rsqrtf(var + 1e-5f);
  AO[ro + lane]      = (_Float16)((v0 - mean) * inv * g[lane] + bta[lane]);
  AO[ro + 32 + lane] = (_Float16)((v1 - mean) * inv * g[32 + lane] + bta[32 + lane]);
}

__global__ void select_kernel(const _Float16* __restrict__ AO, const _Float16* __restrict__ NE,
                              const int* __restrict__ fb, _Float16* __restrict__ EVo, int total) {
  int i = blockIdx.x * blockDim.x + threadIdx.x;
  if (i >= total) return;
  int m = i >> 6;
  EVo[i] = (fb[m] > 0) ? AO[i] : NE[i];
}

__global__ void gather_score_kernel(const int* __restrict__ uid, const int* __restrict__ iid,
                                    const int* __restrict__ nid, const int* __restrict__ ts,
                                    const float* __restrict__ UE, const float* __restrict__ IE,
                                    const float* __restrict__ TE,
                                    const _Float16* __restrict__ ORV,
                                    _Float16* __restrict__ AS1, _Float16* __restrict__ AS2,
                                    _Float16* __restrict__ AS3) {
  int i = blockIdx.x * blockDim.x + threadIdx.x;
  if (i >= MS2 * 64) return;
  int e = i & 63, r = i >> 6, b = r >> 1, j = r & 1;
  int item = j ? nid[b] : iid[b];
  size_t row = (size_t)r * 128;
  AS1[row + e]      = (_Float16)UE[(size_t)uid[b] * 64 + e];
  AS1[row + 64 + e] = (_Float16)IE[(size_t)item * 64 + e];
  AS2[row + 64 + e] = (_Float16)TE[(size_t)ts[b] * 64 + e];
  AS3[row + e]      = ORV[(size_t)b * 64 + e];
}

__global__ void cosine_kernel(const float* __restrict__ EXPR, const float* __restrict__ tv,
                              float* __restrict__ out, int n) {
  int i = blockIdx.x * blockDim.x + threadIdx.x;
  if (i >= n) return;
  float dot = 0.0f, nn = 0.0f, tn = 0.0f;
  for (int e = 0; e < 64; ++e) {
    float x = EXPR[(size_t)i * 64 + e];
    float t = tv[e];
    dot += x * t; nn += x * x; tn += t * t;
  }
  out[i] = 10.0f * dot / (sqrtf(nn) * sqrtf(tn) + 1e-8f);
}

// ---------------- host side ----------------
static void gemm_launch(hipStream_t stream, const _Float16* A, int lda, const _Float16* WT,
                        const float* bias, _Float16* Ch, float* Cf, int ldc,
                        int M, int N, int K, int relu) {
  int tiles = (M >> 5) * (N >> 6);               // 32x64 block per wave
  int blocks = (tiles + 7) >> 3;                 // 8 waves per 256-thread block
  wmma_gemm_kernel<<<dim3(blocks), dim3(256), 0, stream>>>(A, lda, WT, bias, Ch, Cf, ldc,
                                                           M, N, K, relu);
}

extern "C" void kernel_launch(void* const* d_in, const int* in_sizes, int n_in,
                              void* d_out, int out_size, void* d_ws, size_t ws_size,
                              hipStream_t stream) {
  (void)in_sizes; (void)n_in; (void)out_size; (void)ws_size;
  const int* user_ids  = (const int*)d_in[0];
  const int* item_ids  = (const int*)d_in[1];
  const int* histories = (const int*)d_in[2];
  const int* feedbacks = (const int*)d_in[3];
  const int* neg_ids   = (const int*)d_in[4];
  const int* timestamp = (const int*)d_in[5];
  const int* hist_ts   = (const int*)d_in[6];
  const float* user_emb = (const float*)d_in[7];
  const float* item_emb = (const float*)d_in[8];
  const float* time_emb = (const float*)d_in[9];
  const float* true_vec = (const float*)d_in[10];
  const float* not1_w = (const float*)d_in[11]; const float* not1_b = (const float*)d_in[12];
  const float* not2_w = (const float*)d_in[13]; const float* not2_b = (const float*)d_in[14];
  const float* or1_w  = (const float*)d_in[15]; const float* or1_b  = (const float*)d_in[16];
  const float* or2_w  = (const float*)d_in[17]; const float* or2_b  = (const float*)d_in[18];
  const float* and1_w = (const float*)d_in[19]; const float* and1_b = (const float*)d_in[20];
  const float* and2_w = (const float*)d_in[21]; const float* and2_b = (const float*)d_in[22];
  const float* enc1_w = (const float*)d_in[23]; const float* enc1_b = (const float*)d_in[24];
  const float* enc2_w = (const float*)d_in[25]; const float* enc2_b = (const float*)d_in[26];
  const float* tenc1_w = (const float*)d_in[27]; const float* tenc1_b = (const float*)d_in[28];
  const float* tenc2_w = (const float*)d_in[29]; const float* tenc2_b = (const float*)d_in[30];
  const float* WQ  = (const float*)d_in[31]; const float* bQ  = (const float*)d_in[32];
  const float* WK  = (const float*)d_in[33]; const float* bK  = (const float*)d_in[34];
  const float* WV  = (const float*)d_in[35]; const float* bV  = (const float*)d_in[36];
  const float* WQP = (const float*)d_in[37]; const float* bQP = (const float*)d_in[38];
  const float* WKP = (const float*)d_in[39]; const float* bKP = (const float*)d_in[40];
  const float* Wql = (const float*)d_in[41]; const float* bql = (const float*)d_in[42];
  const float* ln_g = (const float*)d_in[43]; const float* ln_b = (const float*)d_in[44];
  float* out = (float*)d_out;

  char* ws = (char*)d_ws;
  size_t off = 0;
  auto alloc = [&](size_t bytes) -> char* {
    char* p = ws + off;
    off = (off + bytes + 255) & ~(size_t)255;
    return p;
  };
  // f16 transposed weight pool
  _Float16* enc1T  = (_Float16*)alloc(64 * 128 * 2);
  _Float16* enc2T  = (_Float16*)alloc(64 * 64 * 2);
  _Float16* tenc1T = (_Float16*)alloc(64 * 128 * 2);
  _Float16* tenc2T = (_Float16*)alloc(64 * 64 * 2);
  _Float16* not1T  = (_Float16*)alloc(64 * 64 * 2);
  _Float16* not2T  = (_Float16*)alloc(64 * 64 * 2);
  _Float16* or1T   = (_Float16*)alloc(64 * 128 * 2);
  _Float16* or2T   = (_Float16*)alloc(64 * 64 * 2);
  _Float16* and1T  = (_Float16*)alloc(64 * 128 * 2);
  _Float16* and2T  = (_Float16*)alloc(64 * 64 * 2);
  _Float16* wqcT   = (_Float16*)alloc(512 * 128 * 2);
  _Float16* wkcT   = (_Float16*)alloc(512 * 128 * 2);
  _Float16* wvcT   = (_Float16*)alloc(256 * 128 * 2);
  _Float16* wqlT   = (_Float16*)alloc(64 * 256 * 2);
  float*    bqc    = (float*)alloc(512 * 4);
  float*    bkc    = (float*)alloc(512 * 4);
  // activations
  _Float16* A1  = (_Float16*)alloc((size_t)MH * 128 * 2);  // [u|item], later reused as EV
  _Float16* H1  = (_Float16*)alloc((size_t)MH * 64 * 2);
  _Float16* A2  = (_Float16*)alloc((size_t)MH * 128 * 2);  // [hist_ev|ht]
  _Float16* XH  = (_Float16*)alloc((size_t)MH * 128 * 2);  // [hist_fused|ht]
  _Float16* QC  = (_Float16*)alloc((size_t)MH * 512 * 2);
  _Float16* KC  = (_Float16*)alloc((size_t)MH * 512 * 2);
  _Float16* Vb  = (_Float16*)alloc((size_t)MH * 256 * 2);
  _Float16* CT  = (_Float16*)alloc((size_t)MH * 256 * 2);
  float*    O32 = (float*)alloc((size_t)MH * 64 * 4);
  _Float16* AO  = (_Float16*)alloc((size_t)MH * 64 * 2);
  _Float16* NE  = (_Float16*)alloc((size_t)MH * 64 * 2);
  _Float16* EVb = A1;                                      // alias: A1 dead by then
  _Float16* ORV = (_Float16*)alloc((size_t)BATCH * 64 * 2);
  _Float16* AS1 = (_Float16*)alloc((size_t)MS2 * 128 * 2);
  _Float16* HS  = (_Float16*)alloc((size_t)MS2 * 64 * 2);
  _Float16* AS2 = (_Float16*)alloc((size_t)MS2 * 128 * 2);
  _Float16* AS3 = (_Float16*)alloc((size_t)MS2 * 128 * 2);
  float*    EXPR = (float*)alloc((size_t)MS2 * 64 * 4);

  auto tg = [](int n) { return dim3((unsigned)((n + 255) / 256)); };

  // weight prep
  transpose_w_kernel<<<tg(128 * 64), 256, 0, stream>>>(enc1_w,  enc1T, 128, 64);
  transpose_w_kernel<<<tg(64 * 64),  256, 0, stream>>>(enc2_w,  enc2T, 64, 64);
  transpose_w_kernel<<<tg(128 * 64), 256, 0, stream>>>(tenc1_w, tenc1T, 128, 64);
  transpose_w_kernel<<<tg(64 * 64),  256, 0, stream>>>(tenc2_w, tenc2T, 64, 64);
  transpose_w_kernel<<<tg(64 * 64),  256, 0, stream>>>(not1_w,  not1T, 64, 64);
  transpose_w_kernel<<<tg(64 * 64),  256, 0, stream>>>(not2_w,  not2T, 64, 64);
  transpose_w_kernel<<<tg(128 * 64), 256, 0, stream>>>(or1_w,   or1T, 128, 64);
  transpose_w_kernel<<<tg(64 * 64),  256, 0, stream>>>(or2_w,   or2T, 64, 64);
  transpose_w_kernel<<<tg(128 * 64), 256, 0, stream>>>(and1_w,  and1T, 128, 64);
  transpose_w_kernel<<<tg(64 * 64),  256, 0, stream>>>(and2_w,  and2T, 64, 64);
  transpose_w_kernel<<<tg(256 * 64), 256, 0, stream>>>(Wql,     wqlT, 256, 64);
  build_qkc_kernel<<<tg(512 * 128), 256, 0, stream>>>(WQ, WQP, bQ, bQP, wqcT, bqc);
  build_qkc_kernel<<<tg(512 * 128), 256, 0, stream>>>(WK, WKP, bK, bKP, wkcT, bkc);
  build_vc_kernel<<<tg(256 * 128), 256, 0, stream>>>(WV, wvcT);

  // history pipeline
  gather_hist_kernel<<<tg(MH * 64), 256, 0, stream>>>(user_ids, histories, hist_ts,
                                                      user_emb, item_emb, time_emb,
                                                      A1, A2, XH);
  gemm_launch(stream, A1, 128, enc1T,  enc1_b,  H1, nullptr, 64,  MH, 64, 128, 1);
  gemm_launch(stream, H1, 64,  enc2T,  enc2_b,  A2, nullptr, 128, MH, 64, 64, 0);
  gemm_launch(stream, A2, 128, tenc1T, tenc1_b, H1, nullptr, 64,  MH, 64, 128, 1);
  gemm_launch(stream, H1, 64,  tenc2T, tenc2_b, XH, nullptr, 128, MH, 64, 64, 0);
  gemm_launch(stream, XH, 128, wqcT,   bqc,     QC, nullptr, 512, MH, 512, 128, 0);
  gemm_launch(stream, XH, 128, wkcT,   bkc,     KC, nullptr, 512, MH, 512, 128, 0);
  gemm_launch(stream, XH, 128, wvcT,   bV,      Vb, nullptr, 256, MH, 256, 128, 0);
  attn_fused_kernel<<<dim3(BATCH * NHEAD), 256, ATTN_SMEM, stream>>>(QC, KC, Vb, CT);
  gemm_launch(stream, CT, 256, wqlT,   bql, nullptr, O32, 64, MH, 64, 256, 0);
  ln_kernel<<<dim3(MH / 8), 256, 0, stream>>>(O32, XH, ln_g, ln_b, AO, MH);
  gemm_launch(stream, AO, 64, not1T, not1_b, H1, nullptr, 64, MH, 64, 64, 1);
  gemm_launch(stream, H1, 64, not2T, not2_b, NE, nullptr, 64, MH, 64, 64, 0);
  select_kernel<<<tg(MH * 64), 256, 0, stream>>>(AO, NE, feedbacks, EVb, MH * 64);
  orscan_kernel<<<dim3(BATCH / 16), 128, 0, stream>>>(EVb, or1T, or1_b, or2T, or2_b, ORV);

  // scoring pipeline (pos + neg interleaved: row = b*2 + j)
  gather_score_kernel<<<tg(MS2 * 64), 256, 0, stream>>>(user_ids, item_ids, neg_ids,
                                                        timestamp, user_emb, item_emb,
                                                        time_emb, ORV, AS1, AS2, AS3);
  gemm_launch(stream, AS1, 128, enc1T,  enc1_b,  HS, nullptr, 64,  MS2, 64, 128, 1);
  gemm_launch(stream, HS,  64,  enc2T,  enc2_b,  AS2, nullptr, 128, MS2, 64, 64, 0);
  gemm_launch(stream, AS2, 128, tenc1T, tenc1_b, HS, nullptr, 64,  MS2, 64, 128, 1);
  gemm_launch(stream, HS,  64,  tenc2T, tenc2_b, AS3 + 64, nullptr, 128, MS2, 64, 64, 0);
  gemm_launch(stream, AS3, 128, and1T,  and1_b,  HS, nullptr, 64,  MS2, 64, 128, 1);
  gemm_launch(stream, HS,  64,  and2T,  and2_b,  nullptr, EXPR, 64, MS2, 64, 64, 0);
  cosine_kernel<<<tg(MS2), 256, 0, stream>>>(EXPR, true_vec, out, MS2);
}
